// Transformer_4629974745556
// MI455X (gfx1250) — matrix operand order repaired and checked by reading, not compiled, not run on hardware
//
#include <hip/hip_runtime.h>
#include <hip/hip_bf16.h>
#include <math.h>

// ---------------------------------------------------------------------------
// Transformer forward for MI455X (gfx1250, wave32, WMMA).
// - All GEMMs + attention use V_WMMA_F32_16X16X32_F16, fp32 accumulation.
// - Weights pre-converted to f16 once per call (403MB read = ~17us @23.3TB/s);
//   all GEMM/attention operands stream as contiguous f16 (no in-loop cvt).
// - RoPE fused with f16 packing of Q (pre-scaled), K, and V^T per head so
//   every attention fragment load is a contiguous 32B f16 read.
// - Flash attention: row-max via DPP row_ror (no LDS bpermute), row-sum via
//   an extra WMMA against an all-ones B fragment, masked tail split out.
// ---------------------------------------------------------------------------

#define NTOK 2048
#define DMODEL 1024
#define NHEAD 16
#define HDIM 64
#define D3 (3 * DMODEL)
#define NLAYER 6

typedef __attribute__((ext_vector_type(16))) _Float16 v16h;
typedef __attribute__((ext_vector_type(4)))  _Float16 v4h;
typedef __attribute__((ext_vector_type(8)))  float    v8f;
typedef __attribute__((ext_vector_type(4)))  float    v4f;

// ---- WMMA fragment loaders (contiguous k order, same for A and B) ---------------
// A-matrix 16x32: lane l -> row m = l%16; halves j = K[c + j],
//                 c = 16*(l/16).
__device__ __forceinline__ v16h load_a_frag(const _Float16* __restrict__ A,
                                            int ld, int lane) {
  const int m  = lane & 15;
  const int kc = (lane >> 4) << 4;
  const _Float16* p = A + (size_t)m * ld + kc;
  v16h a;
#pragma unroll
  for (int j = 0; j < 8; ++j) a[j] = p[j];
#pragma unroll
  for (int j = 0; j < 8; ++j) a[8 + j] = p[8 + j];
  return a;
}

// B-matrix 32x16 (K x N): lane l -> col n = l%16; halves j = K[kb + j],
// kb = 16*(l/16).  B[k][n] is read as row n of W (contiguous in k).
__device__ __forceinline__ v16h load_b_frag(const _Float16* __restrict__ W,
                                            int ld, int lane) {
  const int n  = lane & 15;
  const int kb = (lane >> 4) << 4;
  const _Float16* p = W + (size_t)n * ld + kb;
  v16h b;
#pragma unroll
  for (int j = 0; j < 16; ++j) b[j] = p[j];
  return b;
}

// fp32 fallback (only used when ws cannot hold f16 weight copies)
__device__ __forceinline__ v16h load_b_frag(const float* __restrict__ W,
                                            int ld, int lane) {
  const int n  = lane & 15;
  const int kb = (lane >> 4) << 4;
  const float* p = W + (size_t)n * ld + kb;
  v16h b;
#pragma unroll
  for (int j = 0; j < 16; ++j) b[j] = (_Float16)p[j];
  return b;
}

#define WMMA_F16(a, b, c) \
  __builtin_amdgcn_wmma_f32_16x16x32_f16(false, (a), false, (b), (short)0, (c), false, false)

// ---- DPP row-rotate max reduction over a 16-lane row ------------------------
template <int CTRL>
__device__ __forceinline__ float row_rot(float x) {
  const int v = __builtin_bit_cast(int, x);
  const int r = __builtin_amdgcn_update_dpp(v, v, CTRL, 0xf, 0xf, false);
  return __builtin_bit_cast(float, r);
}

__device__ __forceinline__ float rowmax16(float x) {
  x = fmaxf(x, row_rot<0x121>(x));  // row_ror:1
  x = fmaxf(x, row_rot<0x122>(x));  // row_ror:2
  x = fmaxf(x, row_rot<0x124>(x));  // row_ror:4
  x = fmaxf(x, row_rot<0x128>(x));  // row_ror:8
  return x;
}

// ---------------------------------------------------------------------------
// fp32 -> f16 conversion (weights), 4 elements/thread, vectorized.
// ---------------------------------------------------------------------------
__global__ __launch_bounds__(256) void cvt_f32_f16_kernel(
    const float* __restrict__ in, _Float16* __restrict__ out) {
  const size_t i = ((size_t)blockIdx.x * blockDim.x + threadIdx.x) * 4;
  const v4f v = *(const v4f*)(in + i);
  v4h h;
#pragma unroll
  for (int j = 0; j < 4; ++j) h[j] = (_Float16)v[j];
  *(v4h*)(out + i) = h;
}

// ---------------------------------------------------------------------------
// RMSNorm: one block per row of x[N, D]; writes f16 for the GEMM A-side.
// ---------------------------------------------------------------------------
__global__ __launch_bounds__(256) void rmsnorm_kernel(
    const float* __restrict__ x, const float* __restrict__ w,
    _Float16* __restrict__ h) {
  __shared__ float red[256];
  const int row = blockIdx.x;
  const float* xr = x + (size_t)row * DMODEL;
  float s = 0.f;
  for (int d = threadIdx.x; d < DMODEL; d += 256) {
    const float v = xr[d];
    s += v * v;
  }
  red[threadIdx.x] = s;
  __syncthreads();
  for (int off = 128; off > 0; off >>= 1) {
    if (threadIdx.x < off) red[threadIdx.x] += red[threadIdx.x + off];
    __syncthreads();
  }
  const float rs = rsqrtf(red[0] * (1.0f / DMODEL) + 1e-6f);
  for (int d = threadIdx.x; d < DMODEL; d += 256)
    h[(size_t)row * DMODEL + d] = (_Float16)(xr[d] * rs * w[d]);
}

// ---------------------------------------------------------------------------
// NT GEMM: out[n,e] = sum_d A[n,d] * W[e,d]  (+ optional residual, fp32 out).
// 256 threads = 8 waves; each wave owns a 32x64 tile (8 WMMAs / 32-k step).
// grid = (E/512, N/32).
// ---------------------------------------------------------------------------
template <bool RES, typename WT>
__global__ __launch_bounds__(256) void gemm_nt_kernel(
    const _Float16* __restrict__ A, const WT* __restrict__ W,
    const float* __restrict__ res, float* __restrict__ out,
    int E, int Kin) {
  const int lane = threadIdx.x & 31;
  const int wave = threadIdx.x >> 5;
  const int row0 = blockIdx.y << 5;
  const int col0 = ((blockIdx.x << 3) + wave) << 6;

  v8f acc[2][4];
#pragma unroll
  for (int rr = 0; rr < 2; ++rr)
#pragma unroll
    for (int t = 0; t < 4; ++t)
#pragma unroll
      for (int r = 0; r < 8; ++r) acc[rr][t][r] = 0.f;

  for (int k0 = 0; k0 < Kin; k0 += 32) {
    const v16h a0 = load_a_frag(A + (size_t)row0 * Kin + k0, Kin, lane);
    const v16h a1 = load_a_frag(A + (size_t)(row0 + 16) * Kin + k0, Kin, lane);
#pragma unroll
    for (int t = 0; t < 4; ++t) {
      const WT* wrow = W + (size_t)(col0 + 16 * t) * Kin + k0;
      __builtin_prefetch(wrow + 128, 0, 1);  // next k-tiles of this col strip
      const v16h b = load_b_frag(wrow, Kin, lane);
      acc[0][t] = WMMA_F16(a0, b, acc[0][t]);
      acc[1][t] = WMMA_F16(a1, b, acc[1][t]);
    }
  }

  // C layout: lane l holds (m = r + 8*(l/16), n = l%16) at VGPR r.
  const int n = lane & 15;
  const int g = lane >> 4;
#pragma unroll
  for (int rr = 0; rr < 2; ++rr)
#pragma unroll
    for (int t = 0; t < 4; ++t)
#pragma unroll
      for (int r = 0; r < 8; ++r) {
        const size_t idx =
            (size_t)(row0 + 16 * rr + r + 8 * g) * E + col0 + 16 * t + n;
        float v = acc[rr][t][r];
        if (RES) v += res[idx];
        out[idx] = v;
      }
}

// ---------------------------------------------------------------------------
// Fused SwiGLU up-projection: a = h @ w1^T (w1: [8D, D]);
// g[n,f] = silu(a[n,f]) * a[n,f+4D].  16x64 tile per wave, output f16.
// grid = (4D/512, N/16).
// ---------------------------------------------------------------------------
template <typename WT>
__global__ __launch_bounds__(256) void mlp_gate_kernel(
    const _Float16* __restrict__ h, const WT* __restrict__ w1,
    _Float16* __restrict__ g_out) {
  const int lane = threadIdx.x & 31;
  const int wave = threadIdx.x >> 5;
  const int row0 = blockIdx.y << 4;
  const int col0 = ((blockIdx.x << 3) + wave) << 6;  // [0, 4096)

  v8f a1[4], a2[4];
#pragma unroll
  for (int t = 0; t < 4; ++t)
#pragma unroll
    for (int r = 0; r < 8; ++r) { a1[t][r] = 0.f; a2[t][r] = 0.f; }

  for (int k0 = 0; k0 < DMODEL; k0 += 32) {
    const v16h a = load_a_frag(h + (size_t)row0 * DMODEL + k0, DMODEL, lane);
#pragma unroll
    for (int t = 0; t < 4; ++t) {
      const WT* w1r = w1 + (size_t)(col0 + 16 * t) * DMODEL + k0;
      const WT* w2r = w1 + (size_t)(4096 + col0 + 16 * t) * DMODEL + k0;
      __builtin_prefetch(w1r + 128, 0, 1);
      __builtin_prefetch(w2r + 128, 0, 1);
      const v16h b1 = load_b_frag(w1r, DMODEL, lane);
      a1[t] = WMMA_F16(a, b1, a1[t]);
      const v16h b2 = load_b_frag(w2r, DMODEL, lane);
      a2[t] = WMMA_F16(a, b2, a2[t]);
    }
  }

  const int n = lane & 15;
  const int g = lane >> 4;
#pragma unroll
  for (int t = 0; t < 4; ++t)
#pragma unroll
    for (int r = 0; r < 8; ++r) {
      const float u   = a1[t][r];
      const float sig = 1.0f / (1.0f + __expf(-u));
      g_out[(size_t)(row0 + r + 8 * g) * 4096 + col0 + 16 * t + n] =
          (_Float16)(u * sig * a2[t][r]);
    }
}

// ---------------------------------------------------------------------------
// Fused RoPE + f16 pack for attention:
//   qh[h][n][d] = rope(q)[n,h,d] * (1/sqrt(K))   (f16, [H][N][64])
//   kh[h][n][d] = rope(k)[n,h,d]                 (f16, [H][N][64])
//   vT[h][d][n] = v[n,h,d]                       (f16, [H][64][N])
// ---------------------------------------------------------------------------
__global__ __launch_bounds__(256) void rope_pack_kernel(
    const float* __restrict__ qkv, _Float16* __restrict__ qh,
    _Float16* __restrict__ kh, _Float16* __restrict__ vT) {
  const int idx = blockIdx.x * blockDim.x + threadIdx.x;  // N*H*32
  const int i  = idx & 31;
  const int hh = (idx >> 5) & (NHEAD - 1);
  const int nn = idx >> 9;
  const float freq = __expf(-(float)(2 * i) * (9.210340371976184f / 64.0f));
  float s, c;
  __sincosf((float)nn * freq, &s, &c);

  const float* q = qkv + (size_t)nn * D3 + hh * HDIM;
  const float* k = q + DMODEL;
  const float* v = q + 2 * DMODEL;

  const size_t hb = (size_t)hh * NTOK * HDIM;  // head base (row-major Q/K)
  _Float16* qo = qh + hb + (size_t)nn * HDIM;
  _Float16* ko = kh + hb + (size_t)nn * HDIM;

  const float q0 = q[i], q1 = q[i + 32];
  qo[i]      = (_Float16)((q0 * c - q1 * s) * 0.125f);
  qo[i + 32] = (_Float16)((q1 * c + q0 * s) * 0.125f);
  const float k0 = k[i], k1 = k[i + 32];
  ko[i]      = (_Float16)(k0 * c - k1 * s);
  ko[i + 32] = (_Float16)(k1 * c + k0 * s);

  _Float16* vo = vT + hb;  // [64][N] within head
  vo[(size_t)i * NTOK + nn]        = (_Float16)v[i];
  vo[(size_t)(i + 32) * NTOK + nn] = (_Float16)v[i + 32];
}

// ---------------------------------------------------------------------------
// Flash attention, causal. 1 wave per (head, 16-query block).
// Key blocks of 32: QK^T = 4 WMMAs, PV = 4 WMMAs, row-sum = 1 WMMA vs ones.
// Row-max via DPP row_ror (pure VALU). P bounced through LDS (C->A layout).
// Unmasked bulk loop + masked tail (<=1 block).
// ---------------------------------------------------------------------------
__global__ __launch_bounds__(32) void attn_kernel(
    const _Float16* __restrict__ qh, const _Float16* __restrict__ kh,
    const _Float16* __restrict__ vT, _Float16* __restrict__ o_out) {
  const int lane = threadIdx.x & 31;
  const int head = blockIdx.x;
  const int r0   = blockIdx.y << 4;

  const size_t hb = (size_t)head * NTOK * HDIM;
  const _Float16* qbase = qh + hb;  // [N][64]
  const _Float16* kbase = kh + hb;  // [N][64]
  const _Float16* vbase = vT + hb;  // [64][N]

  __shared__ _Float16 Pl[16][40];  // 16x32 P tile, padded

  // Q fragments (scale already folded in by rope_pack).
  v16h qf[2];
#pragma unroll
  for (int half = 0; half < 2; ++half)
    qf[half] = load_a_frag(qbase + (size_t)r0 * HDIM + half * 32, HDIM, lane);

  // all-ones B fragment: one extra WMMA computes row sums of P.
  v16h onesb;
#pragma unroll
  for (int j = 0; j < 16; ++j) onesb[j] = (_Float16)1.0f;

  v8f o[4];
  float m_i[8], l_i[8];
#pragma unroll
  for (int t = 0; t < 4; ++t)
#pragma unroll
    for (int r = 0; r < 8; ++r) o[t][r] = 0.f;
#pragma unroll
  for (int r = 0; r < 8; ++r) { m_i[r] = -3.0e38f; l_i[r] = 0.f; }

  const int n = lane & 15;
  const int g = lane >> 4;

  auto process_block = [&](int j0, bool domask) {
    v8f s0, s1;
#pragma unroll
    for (int r = 0; r < 8; ++r) { s0[r] = 0.f; s1[r] = 0.f; }
#pragma unroll
    for (int half = 0; half < 2; ++half) {
      const v16h kb0 =
          load_b_frag(kbase + (size_t)j0 * HDIM + half * 32, HDIM, lane);
      s0 = WMMA_F16(qf[half], kb0, s0);
      const v16h kb1 =
          load_b_frag(kbase + (size_t)(j0 + 16) * HDIM + half * 32, HDIM, lane);
      s1 = WMMA_F16(qf[half], kb1, s1);
    }

    if (domask) {
#pragma unroll
      for (int r = 0; r < 8; ++r) {
        const int qrow = r0 + r + 8 * g;
        if (j0 + n      > qrow) s0[r] = -3.0e38f;
        if (j0 + 16 + n > qrow) s1[r] = -3.0e38f;
      }
    }

    // online softmax: row max via DPP rotations (no LDS traffic)
    float alpha[8];
#pragma unroll
    for (int r = 0; r < 8; ++r) {
      const float v = rowmax16(fmaxf(s0[r], s1[r]));
      const float mnew = fmaxf(m_i[r], v);
      alpha[r] = __expf(m_i[r] - mnew);
      s0[r] = __expf(s0[r] - mnew);
      s1[r] = __expf(s1[r] - mnew);
      m_i[r] = mnew;
    }
#pragma unroll
    for (int t = 0; t < 4; ++t)
#pragma unroll
      for (int r = 0; r < 8; ++r) o[t][r] *= alpha[r];

    // P: C-frag layout -> LDS -> A-frag layout
    __syncthreads();
#pragma unroll
    for (int r = 0; r < 8; ++r) {
      Pl[r + 8 * g][n]      = (_Float16)s0[r];
      Pl[r + 8 * g][n + 16] = (_Float16)s1[r];
    }
    __syncthreads();

    v16h pf;
    {
      const int m  = lane & 15;
      const int kc = (lane >> 4) << 4;
#pragma unroll
      for (int j = 0; j < 8; ++j) pf[j] = Pl[m][kc + j];
#pragma unroll
      for (int j = 0; j < 8; ++j) pf[8 + j] = Pl[m][kc + 8 + j];
    }

    // O += P V : B-frag rows are contiguous in vT ([64][N])
#pragma unroll
    for (int t = 0; t < 4; ++t) {
      const v16h vf =
          load_b_frag(vbase + (size_t)(16 * t) * NTOK + j0, NTOK, lane);
      o[t] = WMMA_F16(pf, vf, o[t]);
    }

    // row sums of P via matrix unit: lsum[r] = sum_k P[r+8g][k] (replicated)
    v8f zc;
#pragma unroll
    for (int r = 0; r < 8; ++r) zc[r] = 0.f;
    const v8f lsum = WMMA_F16(pf, onesb, zc);
#pragma unroll
    for (int r = 0; r < 8; ++r) l_i[r] = l_i[r] * alpha[r] + lsum[r];
  };

  int j0 = 0;
  for (; j0 + 31 <= r0; j0 += 32) process_block(j0, false);  // below diagonal
  for (; j0 <= r0 + 15; j0 += 32) process_block(j0, true);   // masked tail

  // normalize + write f16 [N, H*K]
#pragma unroll
  for (int t = 0; t < 4; ++t)
#pragma unroll
    for (int r = 0; r < 8; ++r)
      o_out[(size_t)(r0 + r + 8 * g) * DMODEL + head * HDIM + 16 * t + n] =
          (_Float16)(o[t][r] / l_i[r]);
}

// ---------------------------------------------------------------------------
// Host orchestration (graph-capture safe: only async launches + d2d copy).
// ---------------------------------------------------------------------------
extern "C" void kernel_launch(void* const* d_in, const int* in_sizes, int n_in,
                              void* d_out, int out_size, void* d_ws,
                              size_t ws_size, hipStream_t stream) {
  const float* x_in  = (const float*)d_in[0];
  const float* qkv_w = (const float*)d_in[1];
  const float* out_w = (const float*)d_in[2];
  const float* w1    = (const float*)d_in[3];
  const float* w2    = (const float*)d_in[4];
  const float* n1w   = (const float*)d_in[5];
  const float* n2w   = (const float*)d_in[6];

  const size_t NX = (size_t)NTOK * DMODEL;  // 2M elements
  const size_t QKVW_N = (size_t)NLAYER * 3 * DMODEL * DMODEL;  // 18.9M
  const size_t OUTW_N = (size_t)NLAYER * DMODEL * DMODEL;      //  6.3M
  const size_t W1_N   = (size_t)NLAYER * 8 * DMODEL * DMODEL;  // 50.3M
  const size_t W2_N   = (size_t)NLAYER * 4 * DMODEL * DMODEL;  // 25.2M

  float* ws = (float*)d_ws;
  float*     x_cur = ws;                                 // [N, D]   f32
  float*     qkv   = ws + NX;                            // [N, 3D]  f32
  _Float16*  hbuf  = (_Float16*)(ws + 4 * NX);           // [N, D]   f16
  _Float16*  attn  = (_Float16*)(ws + 4 * NX + NX / 2);  // [N, D]   f16
  _Float16*  gbuf  = (_Float16*)(ws + 5 * NX);           // [N, 4D]  f16
  _Float16*  qhb   = (_Float16*)(ws + 7 * NX);           // [H,N,64] f16
  _Float16*  khb   = (_Float16*)(ws + 7 * NX + NX / 2);  // [H,N,64] f16
  _Float16*  vTb   = (_Float16*)(ws + 8 * NX);           // [H,64,N] f16
  _Float16*  wh    = (_Float16*)(ws + 9 * NX);           // f16 weight copies

  const size_t act_bytes = 9 * NX * sizeof(float);
  const size_t w_total   = QKVW_N + OUTW_N + W1_N + W2_N;  // 100.7M halves
  const bool use_f16w = ws_size >= act_bytes + w_total * sizeof(_Float16);

  _Float16* qkv_wh = wh;
  _Float16* out_wh = qkv_wh + QKVW_N;
  _Float16* w1h    = out_wh + OUTW_N;
  _Float16* w2h    = w1h + W1_N;

  hipMemcpyAsync(x_cur, x_in, NX * sizeof(float), hipMemcpyDeviceToDevice,
                 stream);

  if (use_f16w) {
    cvt_f32_f16_kernel<<<QKVW_N / 1024, 256, 0, stream>>>(qkv_w, qkv_wh);
    cvt_f32_f16_kernel<<<OUTW_N / 1024, 256, 0, stream>>>(out_w, out_wh);
    cvt_f32_f16_kernel<<<W1_N / 1024, 256, 0, stream>>>(w1, w1h);
    cvt_f32_f16_kernel<<<W2_N / 1024, 256, 0, stream>>>(w2, w2h);
  }

  for (int i = 0; i < NLAYER; ++i) {
    const size_t qw_off = (size_t)i * 3 * DMODEL * DMODEL;
    const size_t ow_off = (size_t)i * DMODEL * DMODEL;
    const size_t w1_off = (size_t)i * 8 * DMODEL * DMODEL;
    const size_t w2_off = (size_t)i * DMODEL * 4 * DMODEL;

    rmsnorm_kernel<<<NTOK, 256, 0, stream>>>(x_cur, n1w + i * DMODEL, hbuf);
    if (use_f16w)
      gemm_nt_kernel<false, _Float16><<<dim3(6, NTOK / 32), 256, 0, stream>>>(
          hbuf, qkv_wh + qw_off, nullptr, qkv, 3 * DMODEL, DMODEL);
    else
      gemm_nt_kernel<false, float><<<dim3(6, NTOK / 32), 256, 0, stream>>>(
          hbuf, qkv_w + qw_off, nullptr, qkv, 3 * DMODEL, DMODEL);

    rope_pack_kernel<<<(NTOK * NHEAD * 32) / 256, 256, 0, stream>>>(
        qkv, qhb, khb, vTb);
    attn_kernel<<<dim3(NHEAD, NTOK / 16), 32, 0, stream>>>(qhb, khb, vTb,
                                                           attn);

    if (use_f16w)
      gemm_nt_kernel<true, _Float16><<<dim3(2, NTOK / 32), 256, 0, stream>>>(
          attn, out_wh + ow_off, x_cur, x_cur, DMODEL, DMODEL);
    else
      gemm_nt_kernel<true, float><<<dim3(2, NTOK / 32), 256, 0, stream>>>(
          attn, out_w + ow_off, x_cur, x_cur, DMODEL, DMODEL);

    rmsnorm_kernel<<<NTOK, 256, 0, stream>>>(x_cur, n2w + i * DMODEL, hbuf);

    if (use_f16w)
      mlp_gate_kernel<_Float16><<<dim3(8, NTOK / 16), 256, 0, stream>>>(
          hbuf, w1h + w1_off, gbuf);
    else
      mlp_gate_kernel<float><<<dim3(8, NTOK / 16), 256, 0, stream>>>(
          hbuf, w1 + w1_off, gbuf);

    float* dst = (i == NLAYER - 1) ? (float*)d_out : x_cur;
    if (use_f16w)
      gemm_nt_kernel<true, _Float16><<<dim3(2, NTOK / 32), 256, 0, stream>>>(
          gbuf, w2h + w2_off, x_cur, dst, DMODEL, 4 * DMODEL);
    else
      gemm_nt_kernel<true, float><<<dim3(2, NTOK / 32), 256, 0, stream>>>(
          gbuf, w2 + w2_off, x_cur, dst, DMODEL, 4 * DMODEL);
  }
}
